// GCN_1949915153217
// MI455X (gfx1250) — matrix operand order repaired and checked by reading, not compiled, not run on hardware
//
#include <hip/hip_runtime.h>
#include <hip/hip_bf16.h>

typedef __bf16 bf16_t;
typedef __attribute__((ext_vector_type(16))) __bf16 v16bf;
typedef __attribute__((ext_vector_type(8)))  float  v8f;
typedef __attribute__((ext_vector_type(4)))  unsigned int u32x4;
typedef __attribute__((ext_vector_type(4)))  int i32x4;
typedef __attribute__((ext_vector_type(8)))  int i32x8;

#define NNODES 16384
#define FIN    128

static __device__ __forceinline__ v8f wmma_bf16(v16bf a, v16bf b, v8f c) {
  return __builtin_amdgcn_wmma_f32_16x16x32_bf16(false, a, false, b, (short)0, c, false, false);
}

union Frag  { uint4 q[2]; v16bf v; };
union Pack8 { bf16_t h[8]; uint4 q; };

static __device__ __forceinline__ unsigned lds_off(const void* p) {
  return (unsigned)(size_t)p;    // flat LDS address truncates to LDS byte offset
}

// ---------------------------------------------------------------------------
// TDM: DMA a 2D tile (rows x width bf16, row stride in elems) into LDS with
// 16B padding per row (pad_code: 5 -> pad every 256B, 6 -> every 512B), so the
// LDS row stride becomes width+8 halves. Falls back to a cooperative copy.
// ---------------------------------------------------------------------------
#if __has_builtin(__builtin_amdgcn_tensor_load_to_lds)
static __device__ __forceinline__ void tdm_load_2d(unsigned lds, const bf16_t* g,
                                                   unsigned width, unsigned rows,
                                                   unsigned stride, int pad_code) {
  unsigned long long ga = (unsigned long long)(size_t)g;
  u32x4 g0 = { 1u,                                  // count=1 (valid user D#)
               lds,                                 // lds_addr [63:32]
               (unsigned)ga,                        // global_addr lo
               (unsigned)((ga >> 32) & 0x1ffffffull) | 0x80000000u }; // hi | type=2
  unsigned w0 = (1u << 16)                          // data_size = 2 bytes
              | (1u << 20)                          // pad_enable
              | ((unsigned)pad_code << 22)          // pad_interval
              | (3u << 25);                         // pad_amount = 4 DWORDs (16B)
  i32x8 g1 = { (int)w0,
               (int)((stride & 0xffffu) << 16),                 // tensor_dim0 lo16
               (int)((stride >> 16) | (rows << 16)),            // tensor_dim0 hi | dim1 lo
               (int)(width << 16),                              // dim1 hi(0) | tile_dim0
               (int)rows,                                       // tile_dim1 | tile_dim2=0
               (int)stride,                                     // tensor_dim0_stride lo32
               0, 0 };
  i32x4 z = { 0, 0, 0, 0 };
#if __clang_major__ >= 23
  i32x8 z8 = { 0, 0, 0, 0, 0, 0, 0, 0 };
  __builtin_amdgcn_tensor_load_to_lds(g0, g1, z, z, z8, 0);
#else
  __builtin_amdgcn_tensor_load_to_lds(g0, g1, z, z, 0);
#endif
}
#define HAVE_TDM 1
#else
#define HAVE_TDM 0
#endif

static __device__ __forceinline__ void coop_load_2d(bf16_t* lds, const bf16_t* g,
                                                    int width, int rows,
                                                    long long stride, int KP) {
  int vpr = width >> 3;                              // uint4s per row
  for (int i = threadIdx.x; i < rows * vpr; i += 256) {
    int r = i / vpr, c = (i - r * vpr) << 3;
    *(uint4*)&lds[r * KP + c] = *(const uint4*)&g[(long long)r * stride + c];
  }
}

// ---------------------------------------------------------------------------
// Row-normalize x; emit bf16 x and xn (node-major). One wave per row.
// ---------------------------------------------------------------------------
__global__ void rownorm_kernel(const float* __restrict__ x,
                               bf16_t* __restrict__ xb, bf16_t* __restrict__ xn) {
  int row  = blockIdx.x * 8 + (threadIdx.x >> 5);
  int lane = threadIdx.x & 31;
  float4 v = ((const float4*)(x + (size_t)row * FIN))[lane];
  float s = v.x * v.x + v.y * v.y + v.z * v.z + v.w * v.w;
#pragma unroll
  for (int off = 16; off > 0; off >>= 1) s += __shfl_xor(s, off, 32);
  float r = rsqrtf(s + 1e-12f);
  size_t base = (size_t)row * FIN + (size_t)lane * 4;
  xb[base + 0] = (bf16_t)v.x;  xn[base + 0] = (bf16_t)(v.x * r);
  xb[base + 1] = (bf16_t)v.y;  xn[base + 1] = (bf16_t)(v.y * r);
  xb[base + 2] = (bf16_t)v.z;  xn[base + 2] = (bf16_t)(v.z * r);
  xb[base + 3] = (bf16_t)v.w;  xn[base + 3] = (bf16_t)(v.w * r);
}

// ---------------------------------------------------------------------------
// xnT[f][node] = xn[node][f] via 32-node LDS tiles (all uint4 traffic).
// ---------------------------------------------------------------------------
__global__ void transpose_xn_kernel(const bf16_t* __restrict__ xn, bf16_t* __restrict__ xnT) {
  __shared__ bf16_t tile[32 * 136];                  // 32 nodes x 128 feats, +8 pad
  int node0 = blockIdx.x * 32;
  for (int i = threadIdx.x; i < 32 * 16; i += 256) { // 16 uint4 per row
    int r = i >> 4, c = (i & 15) << 3;
    *(uint4*)&tile[r * 136 + c] = *(const uint4*)(xn + (size_t)(node0 + r) * FIN + c);
  }
  __syncthreads();
  for (int i = threadIdx.x; i < 512; i += 256) {     // 128 feats x 4 uint4
    int f = i >> 2, q = (i & 3) << 3;
    Pack8 pk;
#pragma unroll
    for (int j = 0; j < 8; ++j) pk.h[j] = tile[(q + j) * 136 + f];
    *(uint4*)(xnT + (size_t)f * NNODES + node0 + q) = pk.q;
  }
}

// fp32 -> bf16 straight, and transposed (Wt[n][k] = W[k][n], K = 1<<ksh).
__global__ void cvt_kernel(const float* __restrict__ s, bf16_t* __restrict__ d, int n) {
  int i = blockIdx.x * 256 + threadIdx.x;
  if (i < n) d[i] = (bf16_t)s[i];
}
__global__ void cvtT_kernel(const float* __restrict__ s, bf16_t* __restrict__ d,
                            int Cout, int ksh) {
  int i = blockIdx.x * 256 + threadIdx.x;
  int k = i & ((1 << ksh) - 1), nn = i >> ksh;
  d[i] = (bf16_t)s[k * Cout + nn];
}

// Deterministic split-K sum: Tt = bf16( sum_chunks partial ), total = C*128.
__global__ void reduceT_kernel(const float* __restrict__ partial,
                               bf16_t* __restrict__ Tt, int total, int nchunk) {
  int i = blockIdx.x * 256 + threadIdx.x;
  if (i >= total) return;
  float s = 0.f;
  for (int c = 0; c < nchunk; ++c) s += partial[(size_t)c * total + i];
  Tt[i] = (bf16_t)s;
}

// ---------------------------------------------------------------------------
// D[m][n] = sum_k A[m][k] * Brow[n][k];  A: MxK feature-major (small),
// B: node-major rows of length K (big).  Block tile 64(m) x 64(n), 8 waves.
// Outputs: outN/outNf node-major [node][M_ld] (contiguous 16B/32B stores),
// or outF feature-major [m][NNODES] (coalesced b16 stores) for Pt.
// ---------------------------------------------------------------------------
__global__ void gemm_nodes_kernel(const bf16_t* __restrict__ A,
                                  const bf16_t* __restrict__ B,
                                  const float* __restrict__ bias,
                                  bf16_t* __restrict__ outN, float* __restrict__ outNf,
                                  bf16_t* __restrict__ outF,
                                  int K, int M_ld, int act) {
  __shared__ bf16_t As[64 * 264];
  __shared__ bf16_t Bs[64 * 264];
  const int KP = K + 8;
  const int n0 = blockIdx.x * 64;
  const int m0 = blockIdx.y * 64;
#if HAVE_TDM
  if (threadIdx.x < 32) {
    tdm_load_2d(lds_off(As), A + (size_t)m0 * K, K, 64, K, (K == 256) ? 6 : 5);
    tdm_load_2d(lds_off(Bs), B + (size_t)n0 * K, K, 64, K, (K == 256) ? 6 : 5);
#if __has_builtin(__builtin_amdgcn_s_wait_tensorcnt)
    __builtin_amdgcn_s_wait_tensorcnt(0);
#endif
  }
#else
  coop_load_2d(As, A + (size_t)m0 * K, K, 64, K, KP);
  coop_load_2d(Bs, B + (size_t)n0 * K, K, 64, K, KP);
#endif
  __syncthreads();

  const int wave = threadIdx.x >> 5, lane = threadIdx.x & 31;
  const int m_off  = (wave & 3) << 4;
  const int n_half = (wave >> 2) << 5;               // 0 or 32
  const int mA  = lane & 15;
  const int alo = (lane < 16) ? 0 : 16;
  const int kb  = (lane < 16) ? 0 : 16;
  const int n   = lane & 15;
  if (bias) __builtin_prefetch(&bias[m0 + m_off], 0, 3);

  v8f acc[2] = {{}, {}};
  for (int kk = 0; kk < K; kk += 32) {
    Frag a, b0, b1;
    const bf16_t* ar = &As[(m_off + mA) * KP + kk];
    a.q[0]  = *(const uint4*)(ar + alo);
    a.q[1]  = *(const uint4*)(ar + 8 + alo);
    const bf16_t* br0 = &Bs[(n_half + n) * KP + kk + kb];
    const bf16_t* br1 = &Bs[(n_half + 16 + n) * KP + kk + kb];
    b0.q[0] = *(const uint4*)(br0);  b0.q[1] = *(const uint4*)(br0 + 8);
    b1.q[0] = *(const uint4*)(br1);  b1.q[1] = *(const uint4*)(br1 + 8);
    acc[0] = wmma_bf16(a.v, b0.v, acc[0]);
    acc[1] = wmma_bf16(a.v, b1.v, acc[1]);
  }

  const int mo = (lane < 16) ? 0 : 8;
  const int mbase = m0 + m_off + mo;
  float bv[8] = {0, 0, 0, 0, 0, 0, 0, 0};
  if (bias) {
    float4 t0 = *(const float4*)&bias[mbase];
    float4 t1 = *(const float4*)&bias[mbase + 4];
    bv[0] = t0.x; bv[1] = t0.y; bv[2] = t0.z; bv[3] = t0.w;
    bv[4] = t1.x; bv[5] = t1.y; bv[6] = t1.z; bv[7] = t1.w;
  }
#pragma unroll
  for (int j = 0; j < 2; ++j) {
    const int node = n0 + n_half + j * 16 + n;
    float vals[8];
#pragma unroll
    for (int r = 0; r < 8; ++r) {
      float v = acc[j][r] + bv[r];
      if (act) v = (v > 0.f) ? v : 0.25f * v;        // leaky_relu(0.25)
      vals[r] = v;
    }
    if (outN) {
      Pack8 pk;
#pragma unroll
      for (int r = 0; r < 8; ++r) pk.h[r] = (bf16_t)vals[r];
      *(uint4*)(outN + (size_t)node * M_ld + mbase) = pk.q;
    }
    if (outNf) {
      float4* p = (float4*)(outNf + (size_t)node * M_ld + mbase);
      p[0] = make_float4(vals[0], vals[1], vals[2], vals[3]);
      p[1] = make_float4(vals[4], vals[5], vals[6], vals[7]);
    }
    if (outF) {
#pragma unroll
      for (int r = 0; r < 8; ++r)
        outF[(size_t)(mbase + r) * NNODES + node] = (bf16_t)vals[r];  // coalesced over lanes
    }
  }
}

// ---------------------------------------------------------------------------
// Split-K over 256-node chunks: partialT[chunk][n][m] = (xnT @ Pt-rows) tile.
// A = xnT [128 x NNODES], B = Pt [C x NNODES]; B slab (32 x 256) staged in LDS.
// ---------------------------------------------------------------------------
__global__ void gemm_kred_kernel(const bf16_t* __restrict__ At,
                                 const bf16_t* __restrict__ Bt,
                                 float* __restrict__ part, int C) {
  __shared__ bf16_t Bs[32 * 264];
  const int KP = 264;
  const int kbase = blockIdx.x * 256;
  const int n0 = blockIdx.y * 32;
#if HAVE_TDM
  if (threadIdx.x < 32) {
    tdm_load_2d(lds_off(Bs), Bt + (size_t)n0 * NNODES + kbase, 256, 32, NNODES, 6);
#if __has_builtin(__builtin_amdgcn_s_wait_tensorcnt)
    __builtin_amdgcn_s_wait_tensorcnt(0);
#endif
  }
#else
  coop_load_2d(Bs, Bt + (size_t)n0 * NNODES + kbase, 256, 32, NNODES, KP);
#endif
  __syncthreads();

  const int wave = threadIdx.x >> 5, lane = threadIdx.x & 31;
  const int m_off = wave << 4;                       // 8 waves cover M = 128
  const int mA  = lane & 15;
  const int alo = (lane < 16) ? 0 : 16;
  const int kb  = (lane < 16) ? 0 : 16;
  const int n   = lane & 15;
  const bf16_t* arow = At + (size_t)(m_off + mA) * NNODES + kbase;

  v8f acc[2] = {{}, {}};
#pragma unroll
  for (int kk = 0; kk < 256; kk += 32) {
    Frag a, b0, b1;
    a.q[0]  = *(const uint4*)(arow + kk + alo);
    a.q[1]  = *(const uint4*)(arow + kk + 8 + alo);
    const bf16_t* br0 = &Bs[n * KP + kk + kb];
    const bf16_t* br1 = &Bs[(16 + n) * KP + kk + kb];
    b0.q[0] = *(const uint4*)(br0);  b0.q[1] = *(const uint4*)(br0 + 8);
    b1.q[0] = *(const uint4*)(br1);  b1.q[1] = *(const uint4*)(br1 + 8);
    acc[0] = wmma_bf16(a.v, b0.v, acc[0]);
    acc[1] = wmma_bf16(a.v, b1.v, acc[1]);
  }

  const int mo = (lane < 16) ? 0 : 8;
#pragma unroll
  for (int j = 0; j < 2; ++j) {
    float* d = part + ((size_t)blockIdx.x * C + n0 + j * 16 + n) * 128 + m_off + mo;
    ((float4*)d)[0] = make_float4(acc[j][0], acc[j][1], acc[j][2], acc[j][3]);
    ((float4*)d)[1] = make_float4(acc[j][4], acc[j][5], acc[j][6], acc[j][7]);
  }
}

// ---------------------------------------------------------------------------
extern "C" void kernel_launch(void* const* d_in, const int* in_sizes, int n_in,
                              void* d_out, int out_size, void* d_ws, size_t ws_size,
                              hipStream_t stream) {
  (void)in_sizes; (void)n_in; (void)out_size; (void)ws_size;
  const float* x  = (const float*)d_in[0];
  const float* W1 = (const float*)d_in[1];
  const float* b1 = (const float*)d_in[2];
  const float* W2 = (const float*)d_in[3];
  const float* b2 = (const float*)d_in[4];
  const float* W3 = (const float*)d_in[5];
  const float* b3 = (const float*)d_in[6];
  const float* Wc = (const float*)d_in[7];
  const float* bc = (const float*)d_in[8];
  float* out = (float*)d_out;                        // [N*128] out, then [N*128] h3

  char* w = (char*)d_ws;
  auto take = [&](size_t bytes) { void* p = (void*)w; w += (bytes + 255) & ~(size_t)255; return p; };
  bf16_t* xb  = (bf16_t*)take((size_t)NNODES * 128 * 2);
  bf16_t* xn  = (bf16_t*)take((size_t)NNODES * 128 * 2);
  bf16_t* xnT = (bf16_t*)take((size_t)128 * NNODES * 2);
  bf16_t* W1t = (bf16_t*)take(256 * 128 * 2);
  bf16_t* W2t = (bf16_t*)take(128 * 256 * 2);
  bf16_t* W3t = (bf16_t*)take(128 * 128 * 2);
  bf16_t* Wcb = (bf16_t*)take(128 * 128 * 2);
  bf16_t* Pt  = (bf16_t*)take((size_t)256 * NNODES * 2);
  float*  prt = (float*)take((size_t)(NNODES / 256) * 256 * 128 * 4);
  bf16_t* Tt  = (bf16_t*)take(256 * 128 * 2);
  bf16_t* H1b = (bf16_t*)take((size_t)NNODES * 256 * 2);
  bf16_t* H2b = (bf16_t*)take((size_t)NNODES * 128 * 2);
  bf16_t* H3b = (bf16_t*)take((size_t)NNODES * 128 * 2);

  const int NCH = NNODES / 256;                      // 64 split-K chunks
  const dim3 G64x4(NNODES / 64, 4), G64x2(NNODES / 64, 2);

  rownorm_kernel     <<<NNODES / 8, 256, 0, stream>>>(x, xb, xn);
  transpose_xn_kernel<<<NNODES / 32, 256, 0, stream>>>(xn, xnT);
  cvtT_kernel        <<<128, 256, 0, stream>>>(W1, W1t, 256, 7);   // W1t[256][128]
  cvtT_kernel        <<<128, 256, 0, stream>>>(W2, W2t, 128, 8);   // W2t[128][256]
  cvtT_kernel        <<< 64, 256, 0, stream>>>(W3, W3t, 128, 7);   // W3t[128][128]
  cvt_kernel         <<< 64, 256, 0, stream>>>(Wc, Wcb, 128 * 128); // A for classifier

  // ---- layer 1 (C=256): Pt = (x@W1)^T ; T = xn^T@P ; H1 = lrelu(xn@T + b1)
  gemm_nodes_kernel<<<G64x4, 256, 0, stream>>>(W1t, xb, nullptr, nullptr, nullptr, Pt, 128, 1, 0);
  gemm_kred_kernel <<<dim3(NCH, 8), 256, 0, stream>>>(xnT, Pt, prt, 256);
  reduceT_kernel   <<<128, 256, 0, stream>>>(prt, Tt, 256 * 128, NCH);
  gemm_nodes_kernel<<<G64x4, 256, 0, stream>>>(Tt, xn, b1, H1b, nullptr, nullptr, 128, 256, 1);

  // ---- layer 2 (C=128, K of P-GEMM = 256)
  gemm_nodes_kernel<<<G64x2, 256, 0, stream>>>(W2t, H1b, nullptr, nullptr, nullptr, Pt, 256, 1, 0);
  gemm_kred_kernel <<<dim3(NCH, 4), 256, 0, stream>>>(xnT, Pt, prt, 128);
  reduceT_kernel   <<< 64, 256, 0, stream>>>(prt, Tt, 128 * 128, NCH);
  gemm_nodes_kernel<<<G64x2, 256, 0, stream>>>(Tt, xn, b2, H2b, nullptr, nullptr, 128, 128, 1);

  // ---- layer 3 (C=128): h3 -> d_out second half (f32) + bf16 for classifier
  gemm_nodes_kernel<<<G64x2, 256, 0, stream>>>(W3t, H2b, nullptr, nullptr, nullptr, Pt, 128, 1, 0);
  gemm_kred_kernel <<<dim3(NCH, 4), 256, 0, stream>>>(xnT, Pt, prt, 128);
  reduceT_kernel   <<< 64, 256, 0, stream>>>(prt, Tt, 128 * 128, NCH);
  gemm_nodes_kernel<<<G64x2, 256, 0, stream>>>(Tt, xn, b3, H3b, out + (size_t)NNODES * 128, nullptr, 128, 128, 1);

  // ---- classifier: out[n][j] = sum_k Wc[j][k]*h3[n][k] + bc[j]
  gemm_nodes_kernel<<<G64x2, 256, 0, stream>>>(Wcb, H3b, bc, nullptr, out, nullptr, 128, 128, 0);
}
